// Head_36129264894631
// MI455X (gfx1250) — compile-verified, matrix operand order checked by reading, not run on hardware
//
#include <hip/hip_runtime.h>
#include <hip/hip_bf16.h>

// Single-head causal attention, fused flash-style, one block per batch.
// B=1024, T=256, C=384, H=64.  bf16 WMMA (16x16x32) with f32 accumulation.
// This revision doubles LDS-operand reuse: every B-fragment feeds >= 2 WMMAs.

#define C_EMBD 384
#define HEADD  64
#define T_SEQ  256

typedef __attribute__((ext_vector_type(16))) __bf16 v16bf;
typedef __attribute__((ext_vector_type(8)))  __bf16 v8bf;
typedef __attribute__((ext_vector_type(8)))  float  v8f;

__device__ __forceinline__ v8f wmma_bf16(v16bf a, v16bf b, v8f c) {
  // D = A(16x32) * B(32x16) + C, f32 accum
  return __builtin_amdgcn_wmma_f32_16x16x32_bf16(false, a, false, b,
                                                 (short)0, c, false, false);
}

// Load a 16x32 bf16 A-fragment (also used for B via [n][k] storage) from LDS.
// Lanes 0-15: row=lane, k = 0..7 & 16..23 ; lanes 16-31: row=lane-16,
// k = 8..15 & 24..31.  Two ds_load_b128 per lane.
__device__ __forceinline__ v16bf ldsAfrag(const __bf16* base, int stride, int lane) {
  const int r    = lane & 15;
  const int koff = (lane >> 4) << 3;           // 0 or 8
  const __bf16* p = base + r * stride + koff;
  v8bf lo = *(const v8bf*)(p);
  v8bf hi = *(const v8bf*)(p + 16);
  v16bf o;
#pragma unroll
  for (int i = 0; i < 8; ++i) { o[i] = lo[i]; o[i + 8] = hi[i]; }
  return o;
}

// Load a 16x32 A-fragment of fp32 data from global, converting to bf16.
// p points at &x[row*C + c0 + koff]; four global_load_b128 per lane.
__device__ __forceinline__ v16bf globAfragF32(const float* __restrict__ p) {
  float4 f0 = *(const float4*)(p);
  float4 f1 = *(const float4*)(p + 4);
  float4 f2 = *(const float4*)(p + 16);
  float4 f3 = *(const float4*)(p + 20);
  float v[16] = {f0.x, f0.y, f0.z, f0.w, f1.x, f1.y, f1.z, f1.w,
                 f2.x, f2.y, f2.z, f2.w, f3.x, f3.y, f3.z, f3.w};
  v16bf o;
#pragma unroll
  for (int i = 0; i < 16; ++i) o[i] = (__bf16)v[i];
  return o;
}

extern "C" __global__ __launch_bounds__(256)
void head_attn_kernel(const float* __restrict__ x,
                      const float* __restrict__ Wq,
                      const float* __restrict__ Wk,
                      const float* __restrict__ Wv,
                      float* __restrict__ out) {
  extern __shared__ char smem[];
  __bf16* WtQ = (__bf16*)smem;                  // [64][384] bf16 (W^T)
  __bf16* WtK = WtQ + HEADD * C_EMBD;
  __bf16* WtV = WtK + HEADD * C_EMBD;
  __bf16* Kb  = WtV + HEADD * C_EMBD;           // [256][64]  K (row=key, col=c)
  __bf16* Vt  = Kb  + T_SEQ * HEADD;            // [64][256]  V^T (row=h, col=key)
  __bf16* Scr = Vt  + HEADD * T_SEQ;            // [8 waves][16*64] repack scratch

  const int b      = blockIdx.x;
  const int tid    = threadIdx.x;
  const int wave   = tid >> 5;
  const int lane   = tid & 31;
  const int laneLo = lane & 15;
  const int laneHi = lane >> 4;
  const int r0     = wave * 32;                 // this wave's 32 query rows

  // ---- Phase 1: weights -> bf16, transposed [h][c] (contiguous k for B-frags)
  for (int i = tid; i < HEADD * C_EMBD; i += 256) {
    const int c = i >> 6, h = i & 63;           // W stored [c][h]
    WtQ[h * C_EMBD + c] = (__bf16)Wq[i];
    WtK[h * C_EMBD + c] = (__bf16)Wk[i];
    WtV[h * C_EMBD + c] = (__bf16)Wv[i];
  }
  __syncthreads();

  __bf16* wscr = Scr + wave * (16 * 64);
  const float* xb = x + (size_t)b * T_SEQ * C_EMBD;

  // ---- Phase 2: Q/K/V projections, both 16-row tiles at once so each
  // weight B-fragment feeds two WMMAs.
  v8f aq[2][4], ak[2][4], av[2][4];
  {
    v8f z = {};
#pragma unroll
    for (int t = 0; t < 2; ++t)
#pragma unroll
      for (int n = 0; n < 4; ++n) { aq[t][n] = z; ak[t][n] = z; av[t][n] = z; }
  }
  const int xrow = r0 + laneLo;
  const int koff = laneHi << 3;
  for (int c0 = 0; c0 < C_EMBD; c0 += 32) {
    if (c0 + 32 < C_EMBD) {
      __builtin_prefetch(xb + xrow * C_EMBD + c0 + 32 + koff, 0, 0);
      __builtin_prefetch(xb + (xrow + 16) * C_EMBD + c0 + 32 + koff, 0, 0);
    }
    v16bf xa0 = globAfragF32(xb + xrow * C_EMBD + c0 + koff);
    v16bf xa1 = globAfragF32(xb + (xrow + 16) * C_EMBD + c0 + koff);
#pragma unroll
    for (int n = 0; n < 4; ++n) {
      v16bf bq = ldsAfrag(WtQ + (n * 16) * C_EMBD + c0, C_EMBD, lane);
      aq[0][n] = wmma_bf16(xa0, bq, aq[0][n]);
      aq[1][n] = wmma_bf16(xa1, bq, aq[1][n]);
      v16bf bk = ldsAfrag(WtK + (n * 16) * C_EMBD + c0, C_EMBD, lane);
      ak[0][n] = wmma_bf16(xa0, bk, ak[0][n]);
      ak[1][n] = wmma_bf16(xa1, bk, ak[1][n]);
      v16bf bv = ldsAfrag(WtV + (n * 16) * C_EMBD + c0, C_EMBD, lane);
      av[0][n] = wmma_bf16(xa0, bv, av[0][n]);
      av[1][n] = wmma_bf16(xa1, bv, av[1][n]);
    }
  }

  // D-layout element (reg j): row = m0 + j + 8*laneHi, col = n*16 + laneLo
  v16bf qf[2][2];                               // [m-tile][k-chunk of 32]
#pragma unroll
  for (int t = 0; t < 2; ++t) {
    const int m0 = r0 + t * 16;
#pragma unroll
    for (int n = 0; n < 4; ++n) {
      const int h = n * 16 + laneLo;
#pragma unroll
      for (int j = 0; j < 8; ++j) {
        const int r = m0 + j + 8 * laneHi;
        Kb[r * HEADD + h] = (__bf16)ak[t][n][j];
        Vt[h * T_SEQ + r] = (__bf16)av[t][n][j];
        // Q pre-scaled by 1/sqrt(64); staged in wave scratch for A-repack
        wscr[(j + 8 * laneHi) * HEADD + h] = (__bf16)(aq[t][n][j] * 0.125f);
      }
    }
    qf[t][0] = ldsAfrag(wscr,      HEADD, lane);
    qf[t][1] = ldsAfrag(wscr + 32, HEADD, lane);
  }
  __syncthreads();

  // ---- Phase 3: flash attention; both M-tiles share each K/V fragment.
  // r0 is a multiple of 32, so every step j0 (<= r0) is live for both tiles.
  const float L2E = 1.44269504088896f;
  v8f O[2][4];
  float mr[2][8], lr[2][8];
  {
    v8f z = {};
#pragma unroll
    for (int t = 0; t < 2; ++t) {
#pragma unroll
      for (int n = 0; n < 4; ++n) O[t][n] = z;
#pragma unroll
      for (int j = 0; j < 8; ++j) { mr[t][j] = -1e30f; lr[t][j] = 0.f; }
    }
  }

  for (int j0 = 0; j0 <= r0; j0 += 32) {
    // K^T B-fragments for this 32-key step (shared by both M-tiles)
    v16bf bk[2][2];
#pragma unroll
    for (int half = 0; half < 2; ++half) {
      bk[half][0] = ldsAfrag(Kb + (j0 + half * 16) * HEADD + 0,  HEADD, lane);
      bk[half][1] = ldsAfrag(Kb + (j0 + half * 16) * HEADD + 32, HEADD, lane);
    }
#pragma unroll
    for (int t = 0; t < 2; ++t) {
      const int m0 = r0 + t * 16;
      v8f S[2];
      v8f z = {};
#pragma unroll
      for (int half = 0; half < 2; ++half) {
        v8f s = z;
        s = wmma_bf16(qf[t][0], bk[half][0], s);
        s = wmma_bf16(qf[t][1], bk[half][1], s);
        S[half] = s;
      }
      // causal mask
#pragma unroll
      for (int j = 0; j < 8; ++j) {
        const int r  = m0 + j + 8 * laneHi;
        const int k0 = j0 + laneLo;
        if (k0 > r)      S[0][j] = -1e30f;
        if (k0 + 16 > r) S[1][j] = -1e30f;
      }
      // online softmax; D-layout row j lives across 16 lanes -> xor shuffles
#pragma unroll
      for (int j = 0; j < 8; ++j) {
        float mx = fmaxf(S[0][j], S[1][j]);
        mx = fmaxf(mx, __shfl_xor(mx, 1, 32));
        mx = fmaxf(mx, __shfl_xor(mx, 2, 32));
        mx = fmaxf(mx, __shfl_xor(mx, 4, 32));
        mx = fmaxf(mx, __shfl_xor(mx, 8, 32));
        const float mnew  = fmaxf(mr[t][j], mx);
        const float alpha = exp2f((mr[t][j] - mnew) * L2E);
        const float p0 = exp2f((S[0][j] - mnew) * L2E);
        const float p1 = exp2f((S[1][j] - mnew) * L2E);
        S[0][j] = p0; S[1][j] = p1;
        float rs = p0 + p1;
        rs += __shfl_xor(rs, 1, 32);
        rs += __shfl_xor(rs, 2, 32);
        rs += __shfl_xor(rs, 4, 32);
        rs += __shfl_xor(rs, 8, 32);
        lr[t][j] = lr[t][j] * alpha + rs;
        mr[t][j] = mnew;
#pragma unroll
        for (int n = 0; n < 4; ++n) O[t][n][j] *= alpha;
      }
      // stage P_t (16x32) into wave scratch at tile offset
#pragma unroll
      for (int j = 0; j < 8; ++j) {
        const int rloc = j + 8 * laneHi;
        wscr[t * 512 + rloc * 32 + laneLo]      = (__bf16)S[0][j];
        wscr[t * 512 + rloc * 32 + 16 + laneLo] = (__bf16)S[1][j];
      }
    }
    // O += P * V : each V fragment feeds both tiles
    v16bf pf0 = ldsAfrag(wscr,       32, lane);
    v16bf pf1 = ldsAfrag(wscr + 512, 32, lane);
#pragma unroll
    for (int n = 0; n < 4; ++n) {
      v16bf bv = ldsAfrag(Vt + (n * 16) * T_SEQ + j0, T_SEQ, lane);
      O[0][n] = wmma_bf16(pf0, bv, O[0][n]);
      O[1][n] = wmma_bf16(pf1, bv, O[1][n]);
    }
  }

  // ---- Epilogue: divide by row sum, store fp32
#pragma unroll
  for (int t = 0; t < 2; ++t) {
    const int m0 = r0 + t * 16;
    float* ob = out + ((size_t)b * T_SEQ + m0) * HEADD;
#pragma unroll
    for (int j = 0; j < 8; ++j) {
      const float inv = 1.0f / lr[t][j];
#pragma unroll
      for (int n = 0; n < 4; ++n) {
        ob[(j + 8 * laneHi) * HEADD + n * 16 + laneLo] = O[t][n][j] * inv;
      }
    }
  }
}

extern "C" void kernel_launch(void* const* d_in, const int* in_sizes, int n_in,
                              void* d_out, int out_size, void* d_ws, size_t ws_size,
                              hipStream_t stream) {
  const float* x  = (const float*)d_in[0];
  const float* Wq = (const float*)d_in[1];
  const float* Wk = (const float*)d_in[2];
  const float* Wv = (const float*)d_in[3];
  float* out = (float*)d_out;

  // LDS: 3 * W^T (48KB each) + K (32KB) + V^T (32KB) + 8 * 2KB scratch = 224KB
  const size_t smem = (size_t)(3 * HEADD * C_EMBD + T_SEQ * HEADD +
                               HEADD * T_SEQ + 8 * 16 * HEADD) * sizeof(__bf16);
  dim3 grid(1024), block(256);
  head_attn_kernel<<<grid, block, smem, stream>>>(x, Wq, Wk, Wv, out);
}
